// seq2seq_13460427505667
// MI455X (gfx1250) — matrix operand order repaired and checked by reading, not compiled, not run on hardware
//
#include <hip/hip_runtime.h>
#include <hip/hip_bf16.h>
#include <stdint.h>
#include <math.h>

typedef __bf16 bf16;
typedef __attribute__((ext_vector_type(16))) __bf16 bf16x16;
typedef __attribute__((ext_vector_type(8)))  float   f32x8;

#define B_ROWS   512      // batch
#define HID      1024     // hidden
#define KC_H     32       // k-chunks (32 wide) covering h (1024)
#define KC_TOT   33       // + 1 chunk of padded input features
#define NT_H     64       // 16-col tiles across HID
#define MT_B     32       // 16-row tiles across batch
#define NTILE_G  256      // 16-col tiles across 4*HID gate columns
#define T_ENC    336
#define T_DEC    96

// ---------------------------------------------------------------------------
// zero fill
__global__ __launch_bounds__(256) void zero_kernel(uint32_t* p, int n) {
  for (int i = blockIdx.x * 256 + threadIdx.x; i < n; i += gridDim.x * 256) p[i] = 0u;
}

// ---------------------------------------------------------------------------
// Pack combined weights W[k][n] (k: 0..1023 = Whh^T, 1024..1024+Fin-1 = Wih^T,
// rest zero) into B-fragment layout: frag(kc, gtile) = 32 lanes x 16 bf16,
// lane l holds N = gtile*16 + (l&15), K = kc*32 + (l<16?0:8) + s + (s<8?0:8).
__global__ __launch_bounds__(256)
void pack_w_kernel(const float* __restrict__ Whh, const float* __restrict__ Wih,
                   int Fin, bf16* __restrict__ Bfrag) {
  int idx = blockIdx.x * 256 + threadIdx.x;
  if (idx >= KC_TOT * NTILE_G * 32) return;
  int lane = idx & 31;
  int gt   = (idx >> 5) & 255;
  int kc   = idx >> 13;
  int n     = gt * 16 + (lane & 15);
  int kbase = kc * 32 + ((lane < 16) ? 0 : 8);
  bf16x16 v;
#pragma unroll
  for (int s = 0; s < 16; ++s) {
    int k = kbase + s + ((s < 8) ? 0 : 8);
    float w = 0.f;
    if (k < HID)            w = Whh[(size_t)n * HID + k];
    else if (k < HID + Fin) w = Wih[(size_t)n * Fin + (k - HID)];
    v[s] = (bf16)w;
  }
  *(bf16x16*)(Bfrag + (size_t)idx * 16) = v;
}

// A-fragment slot -> chunk-local k mapping (16x32 bf16 A layout)
__device__ __forceinline__ int a_slot_k(int lane, int s) {
  return (lane < 16) ? ((s < 8) ? s : s + 8) : ((s < 8) ? s + 8 : s + 16);
}

// Pack encoder inputs (embed + concat, pad 16 -> 32) into A-fragment layout.
__global__ __launch_bounds__(256)
void pack_x_kernel(const float* __restrict__ x, const float* __restrict__ emb_hour,
                   const float* __restrict__ emb_week, bf16* __restrict__ xfrag) {
  int idx = blockIdx.x * 256 + threadIdx.x;
  if (idx >= T_ENC * MT_B * 32) return;
  int lane = idx & 31;
  int mt   = (idx >> 5) & 31;
  int t    = idx >> 10;
  int b = mt * 16 + (lane & 15);
  const float* xr = x + ((size_t)t * B_ROWS + b) * 16;
  bf16x16 v;
#pragma unroll
  for (int s = 0; s < 16; ++s) {
    int kl = a_slot_k(lane, s);
    float val = 0.f;
    if (kl < 16) {
      if      (kl == 0) val = emb_hour[(int)xr[15]];
      else if (kl == 1) val = emb_week[(int)xr[14]];
      else              val = xr[kl - 2];
    }
    v[s] = (bf16)val;
  }
  *(bf16x16*)(xfrag + (size_t)idx * 16) = v;
}

// Pack decoder teacher-forcing inputs (t==0 -> zeros) into A-fragment layout.
__global__ __launch_bounds__(256)
void pack_trg_kernel(const float* __restrict__ trg, bf16* __restrict__ dfrag) {
  int idx = blockIdx.x * 256 + threadIdx.x;
  if (idx >= T_DEC * MT_B * 32) return;
  int lane = idx & 31;
  int mt   = (idx >> 5) & 31;
  int t    = idx >> 10;
  int b = mt * 16 + (lane & 15);
  bf16x16 v;
#pragma unroll
  for (int s = 0; s < 16; ++s) {
    int kl = a_slot_k(lane, s);
    float val = 0.f;
    if (kl < 2 && t > 0) val = trg[((size_t)t * B_ROWS + b) * 2 + kl];
    v[s] = (bf16)val;
  }
  *(bf16x16*)(dfrag + (size_t)idx * 16) = v;
}

// ---------------------------------------------------------------------------
// Gate fusion + c update for one 16x16 tile; fills the wave's LDS slice with
// bf16(h_new) in row-major (m,n) order for the A-layout transpose.
__device__ __forceinline__
void gate_epilogue(const f32x8& ai, const f32x8& af, const f32x8& ag, const f32x8& ao,
                   float bi, float bff, float bg, float bo,
                   float* __restrict__ cP, bf16* __restrict__ L, int lane) {
  f32x8 cv = *(const f32x8*)cP;
  const int nloc  = lane & 15;
  const int mbase = (lane >> 4) * 8;
#pragma unroll
  for (int r = 0; r < 8; ++r) {
    float gi = ai[r] + bi;
    float gf = af[r] + bff;
    float gg = ag[r] + bg;
    float go = ao[r] + bo;
    float si = 1.f / (1.f + __expf(-gi));
    float sf = 1.f / (1.f + __expf(-gf));
    float so = 1.f / (1.f + __expf(-go));
    float cn = sf * cv[r] + si * tanhf(gg);
    cv[r] = cn;
    L[(mbase + r) * 16 + nloc] = (bf16)(so * tanhf(cn));
  }
  *(f32x8*)cP = cv;
}

// ---------------------------------------------------------------------------
// One LSTM timestep. Grid 128 blocks x 256 threads (8 waves). Each wave owns a
// 32x16 tile of h (2 stacked M-tiles): per K-chunk it loads 2 A-frags + 4
// B-frags and issues 8 WMMAs, so every weight fragment is reused across two
// A fragments (halved L2 weight traffic vs 16x16/wave). Gates are fused
// in-register; new h is emitted in A-fragment layout for the next step via a
// per-wave LDS transpose.
__global__ __launch_bounds__(256)
void lstm_step_kernel(const char*  __restrict__ Bfrag,     // KC_TOT*NTILE_G*1024 B
                      const char*  __restrict__ hfrag_in,  // MT_B*KC_H*1024 B
                      const char*  __restrict__ xfrag_t,   // MT_B*1024 B (this t)
                      char*        __restrict__ hfrag_out,
                      float*       __restrict__ cfrag,     // MT_B*NT_H frags of 32x8 f32
                      const float* __restrict__ bias)      // 4*HID
{
  __shared__ __align__(16) bf16 lds_h[8][2][256];
  const int tid  = threadIdx.x;
  const int lane = tid & 31;
  const int wave = tid >> 5;
  const int bm = blockIdx.x & 7;        // 8 blocks of 64 batch rows
  const int bn = blockIdx.x >> 3;       // 16 blocks of 64 h cols
  const int wm = wave >> 2;             // 0..1 : which 32-row group
  const int wn = wave & 3;              // 0..3 : which 16-col tile
  const int mt0 = bm * 4 + wm * 2;      // 0..31 (and mt0+1)
  const int nt  = bn * 4 + wn;          // 0..63

  f32x8 a00 = {}, a01 = {}, a02 = {}, a03 = {};  // m-tile 0, gates i,f,g,o
  f32x8 a10 = {}, a11 = {}, a12 = {}, a13 = {};  // m-tile 1, gates i,f,g,o

  const char* aP0 = hfrag_in + (size_t)(mt0    ) * KC_H * 1024 + (size_t)lane * 32;
  const char* aP1 = hfrag_in + (size_t)(mt0 + 1) * KC_H * 1024 + (size_t)lane * 32;
  const char* bP  = Bfrag + (size_t)lane * 32;

  for (int kc = 0; kc < KC_H; ++kc) {
    bf16x16 a0 = *(const bf16x16*)(aP0 + (size_t)kc * 1024);
    bf16x16 a1 = *(const bf16x16*)(aP1 + (size_t)kc * 1024);
    const char* bk = bP + (size_t)kc * (NTILE_G * 1024);
    bf16x16 b0 = *(const bf16x16*)(bk + (size_t)(0 * NT_H + nt) * 1024);
    bf16x16 b1 = *(const bf16x16*)(bk + (size_t)(1 * NT_H + nt) * 1024);
    bf16x16 b2 = *(const bf16x16*)(bk + (size_t)(2 * NT_H + nt) * 1024);
    bf16x16 b3 = *(const bf16x16*)(bk + (size_t)(3 * NT_H + nt) * 1024);
    a00 = __builtin_amdgcn_wmma_f32_16x16x32_bf16(false, a0, false, b0, (short)0, a00, false, false);
    a10 = __builtin_amdgcn_wmma_f32_16x16x32_bf16(false, a1, false, b0, (short)0, a10, false, false);
    a01 = __builtin_amdgcn_wmma_f32_16x16x32_bf16(false, a0, false, b1, (short)0, a01, false, false);
    a11 = __builtin_amdgcn_wmma_f32_16x16x32_bf16(false, a1, false, b1, (short)0, a11, false, false);
    a02 = __builtin_amdgcn_wmma_f32_16x16x32_bf16(false, a0, false, b2, (short)0, a02, false, false);
    a12 = __builtin_amdgcn_wmma_f32_16x16x32_bf16(false, a1, false, b2, (short)0, a12, false, false);
    a03 = __builtin_amdgcn_wmma_f32_16x16x32_bf16(false, a0, false, b3, (short)0, a03, false, false);
    a13 = __builtin_amdgcn_wmma_f32_16x16x32_bf16(false, a1, false, b3, (short)0, a13, false, false);
  }
  { // input-feature chunk (kc == 32), A from packed x_t
    bf16x16 a0 = *(const bf16x16*)(xfrag_t + ((size_t)(mt0    ) * 32 + lane) * 32);
    bf16x16 a1 = *(const bf16x16*)(xfrag_t + ((size_t)(mt0 + 1) * 32 + lane) * 32);
    const char* bk = bP + (size_t)KC_H * (NTILE_G * 1024);
    bf16x16 b0 = *(const bf16x16*)(bk + (size_t)(0 * NT_H + nt) * 1024);
    bf16x16 b1 = *(const bf16x16*)(bk + (size_t)(1 * NT_H + nt) * 1024);
    bf16x16 b2 = *(const bf16x16*)(bk + (size_t)(2 * NT_H + nt) * 1024);
    bf16x16 b3 = *(const bf16x16*)(bk + (size_t)(3 * NT_H + nt) * 1024);
    a00 = __builtin_amdgcn_wmma_f32_16x16x32_bf16(false, a0, false, b0, (short)0, a00, false, false);
    a10 = __builtin_amdgcn_wmma_f32_16x16x32_bf16(false, a1, false, b0, (short)0, a10, false, false);
    a01 = __builtin_amdgcn_wmma_f32_16x16x32_bf16(false, a0, false, b1, (short)0, a01, false, false);
    a11 = __builtin_amdgcn_wmma_f32_16x16x32_bf16(false, a1, false, b1, (short)0, a11, false, false);
    a02 = __builtin_amdgcn_wmma_f32_16x16x32_bf16(false, a0, false, b2, (short)0, a02, false, false);
    a12 = __builtin_amdgcn_wmma_f32_16x16x32_bf16(false, a1, false, b2, (short)0, a12, false, false);
    a03 = __builtin_amdgcn_wmma_f32_16x16x32_bf16(false, a0, false, b3, (short)0, a03, false, false);
    a13 = __builtin_amdgcn_wmma_f32_16x16x32_bf16(false, a1, false, b3, (short)0, a13, false, false);
  }

  // Lane holds column n = nt*16 + (lane&15), rows m = (lane>=16)*8 + r.
  const int ncol = nt * 16 + (lane & 15);
  const float bi  = bias[ncol];
  const float bff = bias[HID + ncol];
  const float bg  = bias[2 * HID + ncol];
  const float bo  = bias[3 * HID + ncol];

  float* cP0 = cfrag + ((size_t)((mt0    ) * NT_H + nt) * 32 + lane) * 8;
  float* cP1 = cfrag + ((size_t)((mt0 + 1) * NT_H + nt) * 32 + lane) * 8;
  gate_epilogue(a00, a01, a02, a03, bi, bff, bg, bo, cP0, &lds_h[wave][0][0], lane);
  gate_epilogue(a10, a11, a12, a13, bi, bff, bg, bo, cP1, &lds_h[wave][1][0], lane);
  __syncthreads();

  // Read back row-major (lane = row) and store h fragments for next step.
  const int mrow = lane & 15;
  const int koff = (lane >> 4) * 8;
  uint4 v0 = *(const uint4*)(const void*)(&lds_h[wave][0][0] + mrow * 16 + koff);
  uint4 v1 = *(const uint4*)(const void*)(&lds_h[wave][1][0] + mrow * 16 + koff);
  char* dst0 = hfrag_out + (((size_t)(mt0    ) * KC_H + (nt >> 1)) * 32 + lane) * 32
                         + (size_t)(nt & 1) * 16;
  char* dst1 = hfrag_out + (((size_t)(mt0 + 1) * KC_H + (nt >> 1)) * 32 + lane) * 32
                         + (size_t)(nt & 1) * 16;
  *(uint4*)dst0 = v0;
  *(uint4*)dst1 = v1;
}

// ---------------------------------------------------------------------------
// pred[b][o] = h[b] . W_out[o] + b_out[o], reading h in A-fragment layout.
__global__ __launch_bounds__(256)
void pred_kernel(const char* __restrict__ hfrag, const float* __restrict__ Wout,
                 const float* __restrict__ bout, float* __restrict__ out) {
  int tid = blockIdx.x * 256 + threadIdx.x;
  if (tid >= B_ROWS * 2) return;
  int b = tid >> 1, o = tid & 1;
  int mt = b >> 4, m = b & 15;
  const float* W = Wout + (size_t)o * HID;
  float sum = 0.f;
  for (int kc = 0; kc < KC_H; ++kc) {
#pragma unroll
    for (int half = 0; half < 2; ++half) {
      int lane = m + 16 * half;
      const uint16_t* p = (const uint16_t*)(hfrag + ((size_t)(mt * KC_H + kc) * 32 + lane) * 32);
      int kbase = kc * 32;
#pragma unroll
      for (int s = 0; s < 16; ++s) {
        int k = kbase + ((s < 8) ? (half * 8 + s) : (16 + half * 8 + (s - 8)));
        float hv = __uint_as_float(((uint32_t)p[s]) << 16);
        sum += hv * W[k];
      }
    }
  }
  out[(size_t)b * 2 + o] = sum + bout[o];
}

// ---------------------------------------------------------------------------
extern "C" void kernel_launch(void* const* d_in, const int* in_sizes, int n_in,
                              void* d_out, int out_size, void* d_ws, size_t ws_size,
                              hipStream_t stream) {
  (void)in_sizes; (void)n_in; (void)out_size; (void)ws_size;
  const float* x        = (const float*)d_in[0];
  const float* trg      = (const float*)d_in[1];
  const float* emb_hour = (const float*)d_in[2];
  const float* emb_week = (const float*)d_in[3];
  const float* Wih_e    = (const float*)d_in[4];
  const float* Whh_e    = (const float*)d_in[5];
  const float* b_e      = (const float*)d_in[6];
  const float* Wih_d    = (const float*)d_in[7];
  const float* Whh_d    = (const float*)d_in[8];
  const float* b_d      = (const float*)d_in[9];
  const float* W_out    = (const float*)d_in[10];
  const float* b_out    = (const float*)d_in[11];
  // d_in[12] = isTrain (assumed 1: teacher forcing)

  char* ws = (char*)d_ws;
  const size_t SZ_BFRAG = (size_t)KC_TOT * NTILE_G * 1024;      // 8,650,752
  const size_t SZ_HFRAG = (size_t)MT_B * KC_H * 1024;           // 1,048,576
  const size_t SZ_CFRAG = (size_t)MT_B * NT_H * 1024;           // 2,097,152
  const size_t SZ_XFRAG = (size_t)T_ENC * MT_B * 1024;          // 11,010,048
  char*  Be = ws;
  char*  Bd = Be + SZ_BFRAG;
  char*  hA = Bd + SZ_BFRAG;
  char*  hB = hA + SZ_HFRAG;
  float* cf = (float*)(hB + SZ_HFRAG);
  char*  xf = (char*)cf + SZ_CFRAG;
  char*  df = xf + SZ_XFRAG;

  // Pack weights / inputs, zero initial state.
  pack_w_kernel<<<1056, 256, 0, stream>>>(Whh_e, Wih_e, 16, (bf16*)Be);
  pack_w_kernel<<<1056, 256, 0, stream>>>(Whh_d, Wih_d, 2,  (bf16*)Bd);
  pack_x_kernel<<<1344, 256, 0, stream>>>(x, emb_hour, emb_week, (bf16*)xf);
  pack_trg_kernel<<<384, 256, 0, stream>>>(trg, (bf16*)df);
  zero_kernel<<<512, 256, 0, stream>>>((uint32_t*)hA, (int)(SZ_HFRAG / 4));
  zero_kernel<<<512, 256, 0, stream>>>((uint32_t*)cf, (int)(SZ_CFRAG / 4));

  int s = 0;
  for (int t = 0; t < T_ENC; ++t, ++s) {
    char* hin  = (s & 1) ? hB : hA;
    char* hout = (s & 1) ? hA : hB;
    lstm_step_kernel<<<128, 256, 0, stream>>>(Be, hin, xf + (size_t)t * (MT_B * 1024),
                                              hout, cf, b_e);
  }
  for (int t = 0; t < T_DEC; ++t, ++s) {
    char* hin  = (s & 1) ? hB : hA;
    char* hout = (s & 1) ? hA : hB;
    lstm_step_kernel<<<128, 256, 0, stream>>>(Bd, hin, df + (size_t)t * (MT_B * 1024),
                                              hout, cf, b_d);
    pred_kernel<<<4, 256, 0, stream>>>(hout, W_out, b_out,
                                       (float*)d_out + (size_t)t * (B_ROWS * 2));
  }
}
